// GraphEncoder_20641612824821
// MI455X (gfx1250) — matrix operand order repaired and checked by reading, not compiled, not run on hardware
//
#include <hip/hip_runtime.h>
#include <math.h>

// ---------------------------------------------------------------------------
// GAT encoder for MI455X (gfx1250, wave32, WMMA).
//  * GEMMs: bf16 WMMA (v_wmma_f32_16x16x32_bf16), fp32 accumulate.
//  * Weight tile staged in LDS via GLOBAL_LOAD_ASYNC_TO_LDS_B128 (ASYNCcnt).
//  * Edge softmax / scatter phases fp32 with native global float atomics.
// ---------------------------------------------------------------------------

#define EMB   128
#define HIDD  64
#define HEADS 4
#define NGRAPH 128

typedef __attribute__((ext_vector_type(16))) __bf16 v16bf;
typedef __attribute__((ext_vector_type(8)))  float  v8f;

static __device__ __forceinline__ void atomicMaxF(float* addr, float v) {
  // monotone bit trick: signed max for v>=0, unsigned min for v<0.
  if (v >= 0.0f) {
    atomicMax((int*)addr, __float_as_int(v));
  } else {
    atomicMin((unsigned int*)addr, (unsigned int)__float_as_int(v));
  }
}

// Async copy of 16 bytes global -> LDS (gfx1250 TDM-adjacent path, ASYNCcnt).
static __device__ __forceinline__ void async_g2l_b128(unsigned lds_addr,
                                                      const void* gaddr) {
  asm volatile("global_load_async_to_lds_b128 %0, %1, off"
               :: "v"(lds_addr), "v"((unsigned long long)(size_t)gaddr)
               : "memory");
}
static __device__ __forceinline__ void async_wait0() {
  asm volatile("s_wait_asynccnt 0x0" ::: "memory");
}

// ---------------------------------------------------------------------------
// bf16 WMMA GEMM: C[M,Nn] = A[M,K] @ B[K,Nn]  (fp32 in/out), K compile-time.
// Requires M%16==0, K%32==0, Nn%64==0. Block = 128 threads = 4 waves.
// Block computes a 16-row x 64-col tile; wave w owns columns [w*16, w*16+16).
// The K x 64 weight window is staged in LDS once (async), reused by all waves.
// ---------------------------------------------------------------------------
template <int K>
__global__ __launch_bounds__(128)
void k_gemm_bf16(const float* __restrict__ A, const float* __restrict__ Bm,
                 float* __restrict__ C, int M, int Nn) {
  __shared__ float sB[K * 64];

  const int tid    = threadIdx.x;
  const int wave   = tid >> 5;
  const int lane   = tid & 31;
  const int laneHi = lane >> 4;        // 0 or 1
  const int l16    = lane & 15;
  const int rowBase  = blockIdx.x * 16;
  const int colBase0 = blockIdx.y * 64;

  // ---- stage B tile (K x 64 fp32) into LDS: one b128 async copy per lane ----
  const unsigned sB_base = (unsigned)(size_t)&sB[0];   // low 32b = LDS offset
#pragma unroll 1
  for (int idx = tid; idx < (K * 64) / 4; idx += 128) {  // trip count % 128 == 0
    const int k  = idx >> 4;           // 16 float4 per 64-col row
    const int c4 = (idx & 15) << 2;
    async_g2l_b128(sB_base + (unsigned)(k * 64 + c4) * 4u,
                   Bm + (size_t)k * Nn + colBase0 + c4);
  }
  async_wait0();
  __syncthreads();

  // ---- WMMA main loop ----
  v8f c = {0.f, 0.f, 0.f, 0.f, 0.f, 0.f, 0.f, 0.f};
  const float* arow = A + (size_t)(rowBase + l16) * K;

#pragma unroll
  for (int kb = 0; kb < K; kb += 32) {
    v16bf a, b;
    // A 16x32 bf16 fragment (ISA 7.12.2): lane holds row M=lane%16;
    // lane<16 -> K {kb+0..7, kb+16..23}; lane>=16 -> K {kb+8..15, kb+24..31}.
    const int ka = kb + (laneHi ? 8 : 0);
#pragma unroll
    for (int e = 0; e < 8; ++e) a[e]     = (__bf16)arow[ka + e];
#pragma unroll
    for (int e = 0; e < 8; ++e) a[8 + e] = (__bf16)arow[ka + 16 + e];
    // B 32x16 bf16 fragment: lane holds column N=lane%16, K order same as A.
    const float* bcol = &sB[wave * 16 + l16];
#pragma unroll
    for (int e = 0; e < 8; ++e) b[e]     = (__bf16)bcol[(ka + e) * 64];
#pragma unroll
    for (int e = 0; e < 8; ++e) b[8 + e] = (__bf16)bcol[(ka + 16 + e) * 64];

    c = __builtin_amdgcn_wmma_f32_16x16x32_bf16(
        /*neg_a=*/false, a, /*neg_b=*/false, b,
        /*c_mod=*/(short)0, c, /*reuse_a=*/false, /*reuse_b=*/false);
  }

  // C/D layout: VGPR v, lane -> element (row = v + 8*laneHi, col = lane%16)
  float* cbase = C + (size_t)rowBase * Nn + colBase0 + wave * 16 + l16;
#pragma unroll
  for (int v = 0; v < 8; ++v)
    cbase[(size_t)(v + 8 * laneHi) * Nn] = c[v];
}

// ---------------------------------------------------------------------------
// Per-node attention logits: es[n,h] = <Z[n,h,:], a_s[h,:]>, ed likewise.
// ---------------------------------------------------------------------------
__global__ void k_logits(const float* __restrict__ Z,
                         const float* __restrict__ a_s,
                         const float* __restrict__ a_d,
                         float* __restrict__ es, float* __restrict__ ed,
                         int n, int H, int dch) {
  int t = blockIdx.x * blockDim.x + threadIdx.x;
  if (t >= n * H) return;
  int node = t / H, h = t % H;
  const float* z = Z + (size_t)node * H * dch + (size_t)h * dch;
  float s = 0.f, d = 0.f;
  for (int i = 0; i < dch; ++i) {
    float zv = z[i];
    s += zv * a_s[h * dch + i];
    d += zv * a_d[h * dch + i];
  }
  es[t] = s;
  ed[t] = d;
}

static __device__ __forceinline__ void edge_sd(const int* ei, int E, int e,
                                               int& s, int& d) {
  if (e < E) { s = ei[e]; d = ei[E + e]; }
  else       { s = d = e - E; }          // appended self-loops
}

__global__ void k_edge_max(const int* __restrict__ ei, int E, int n,
                           const float* __restrict__ es,
                           const float* __restrict__ ed,
                           float* __restrict__ m, int H) {
  int e = blockIdx.x * blockDim.x + threadIdx.x;
  if (e >= E + n) return;
  int s, d; edge_sd(ei, E, e, s, d);
  for (int h = 0; h < H; ++h) {
    float x = es[s * H + h] + ed[d * H + h];
    x = x > 0.f ? x : 0.2f * x;          // leaky_relu
    atomicMaxF(&m[d * H + h], x);
  }
}

__global__ void k_edge_exp(const int* __restrict__ ei, int E, int n,
                           const float* __restrict__ es,
                           const float* __restrict__ ed,
                           const float* __restrict__ m,
                           float* __restrict__ ex, float* __restrict__ den,
                           int H) {
  int e = blockIdx.x * blockDim.x + threadIdx.x;
  if (e >= E + n) return;
  int s, d; edge_sd(ei, E, e, s, d);
  for (int h = 0; h < H; ++h) {
    float x = es[s * H + h] + ed[d * H + h];
    x = x > 0.f ? x : 0.2f * x;
    float v = __expf(x - m[d * H + h]);
    ex[(size_t)e * H + h] = v;
    atomicAdd(&den[d * H + h], v);
  }
}

__global__ void k_edge_scatter(const int* __restrict__ ei, int E, int n,
                               const float* __restrict__ Z,
                               const float* __restrict__ ex,
                               const float* __restrict__ den,
                               float* __restrict__ acc, int H, int dch) {
  int t = blockIdx.x * blockDim.x + threadIdx.x;
  if (t >= (E + n) * H) return;
  int e = t / H, h = t % H;
  int s, d; edge_sd(ei, E, e, s, d);
  float alpha = ex[(size_t)e * H + h] / (den[d * H + h] + 1e-16f);
  const float* zs = Z   + (size_t)s * H * dch + (size_t)h * dch;
  float*       ad = acc + (size_t)d * H * dch + (size_t)h * dch;
  for (int i = 0; i < dch; ++i) atomicAdd(&ad[i], alpha * zs[i]);
}

__global__ void k_elu_bias(float* __restrict__ acc,
                           const float* __restrict__ bias, int n, int w) {
  int t = blockIdx.x * blockDim.x + threadIdx.x;
  if (t >= n * w) return;
  float v = acc[t] + bias[t % w];
  acc[t] = v > 0.f ? v : (__expf(v) - 1.f);   // ELU, in place
}

// gate[n] = relu(G1[n,:] + bg1) . Wg2 + bg2
__global__ void k_gate(const float* __restrict__ G1,
                       const float* __restrict__ bg1,
                       const float* __restrict__ Wg2,
                       const float* __restrict__ bg2,
                       float* __restrict__ gate, int n, int dch) {
  int t = blockIdx.x * blockDim.x + threadIdx.x;
  if (t >= n) return;
  float g = 0.f;
  const float* row = G1 + (size_t)t * dch;
  for (int i = 0; i < dch; ++i) {
    float r = row[i] + bg1[i];
    r = r > 0.f ? r : 0.f;
    g += r * Wg2[i];
  }
  gate[t] = g + bg2[0];
}

__global__ void k_bmax(const float* __restrict__ gate,
                       const int* __restrict__ bm,
                       float* __restrict__ gm, int n) {
  int t = blockIdx.x * blockDim.x + threadIdx.x;
  if (t >= n) return;
  atomicMaxF(&gm[bm[t]], gate[t]);
}

__global__ void k_bexp(const float* __restrict__ gate,
                       const int* __restrict__ bm,
                       const float* __restrict__ gm,
                       float* __restrict__ exg, float* __restrict__ gden,
                       int n) {
  int t = blockIdx.x * blockDim.x + threadIdx.x;
  if (t >= n) return;
  float v = __expf(gate[t] - gm[bm[t]]);
  exg[t] = v;
  atomicAdd(&gden[bm[t]], v);
}

__global__ void k_bout(const float* __restrict__ h2,
                       const float* __restrict__ exg,
                       const float* __restrict__ gden,
                       const int* __restrict__ bm,
                       float* __restrict__ out, int n, int dch) {
  int t = blockIdx.x * blockDim.x + threadIdx.x;
  if (t >= n * dch) return;
  int node = t / dch, j = t % dch;
  int g = bm[node];
  float attn = exg[node] / (gden[g] + 1e-16f);
  atomicAdd(&out[g * dch + j], attn * h2[t]);
}

__global__ void k_fill(float* __restrict__ p, float v, size_t cnt) {
  size_t t = (size_t)blockIdx.x * blockDim.x + threadIdx.x;
  if (t < cnt) p[t] = v;
}

static inline unsigned cdiv(size_t a, unsigned b) { return (unsigned)((a + b - 1) / b); }

extern "C" void kernel_launch(void* const* d_in, const int* in_sizes, int n_in,
                              void* d_out, int out_size, void* d_ws, size_t ws_size,
                              hipStream_t stream) {
  const float* x    = (const float*)d_in[0];
  const int*   ei   = (const int*)  d_in[1];
  const int*   bm   = (const int*)  d_in[2];
  const float* W1   = (const float*)d_in[3];
  const float* as1  = (const float*)d_in[4];
  const float* ad1  = (const float*)d_in[5];
  const float* b1   = (const float*)d_in[6];
  const float* W2   = (const float*)d_in[7];
  const float* as2  = (const float*)d_in[8];
  const float* ad2  = (const float*)d_in[9];
  const float* b2   = (const float*)d_in[10];
  const float* Wg1  = (const float*)d_in[11];
  const float* bg1  = (const float*)d_in[12];
  const float* Wg2  = (const float*)d_in[13];
  const float* bg2  = (const float*)d_in[14];
  float* out = (float*)d_out;

  const int N  = in_sizes[0] / EMB;     // 50000
  const int E  = in_sizes[1] / 2;       // 800000
  const int E2 = E + N;
  const int F1 = HEADS * HIDD;          // 256

  // ---- workspace layout (floats) ----
  float* ws = (float*)d_ws;
  size_t o = 0;
  float* Z1  = ws + o;  o += (size_t)N * F1;       // layer-1 z
  float* ACC = ws + o;  o += (size_t)N * F1;       // layer-1 accum -> h1 (in place)
  float* EX1 = ws + o;  o += (size_t)E2 * HEADS;   // layer-1 edge exp
  float* ES  = ws + o;  o += (size_t)N * HEADS;
  float* ED  = ws + o;  o += (size_t)N * HEADS;
  float* MX  = ws + o;  o += (size_t)N * HEADS;
  float* DN  = ws + o;  o += (size_t)N * HEADS;
  // layer-2 / pooling buffers alias the (then-dead) Z1 and EX1 regions
  float* Z2   = Z1;                                // N*HIDD
  float* ACC2 = Z1 + (size_t)N * HIDD;             // N*HIDD -> h2 in place
  float* G1   = Z1 + (size_t)2 * N * HIDD;         // N*HIDD
  float* EX2  = Z1 + (size_t)3 * N * HIDD;         // E2  (<= N*HIDD)
  float* GATE = EX1;                               // N
  float* EXG  = EX1 + N;                           // N
  float* GM   = EX1 + (size_t)2 * N;               // NGRAPH
  float* GDEN = GM + NGRAPH;                       // NGRAPH

  const unsigned TB = 256;
  const float NEG_INF = -INFINITY;

  // ================= Layer 1 (HEADS=4, dch=64) =================
  // Z1 = x @ W1   [N,128] x [128,256]
  k_gemm_bf16<EMB><<<dim3(N / 16, F1 / 64), 128, 0, stream>>>(x, W1, Z1, N, F1);
  k_logits<<<cdiv((size_t)N * HEADS, TB), TB, 0, stream>>>(Z1, as1, ad1, ES, ED, N, HEADS, HIDD);
  k_fill<<<cdiv((size_t)N * HEADS, TB), TB, 0, stream>>>(MX, NEG_INF, (size_t)N * HEADS);
  k_fill<<<cdiv((size_t)N * HEADS, TB), TB, 0, stream>>>(DN, 0.f, (size_t)N * HEADS);
  k_fill<<<cdiv((size_t)N * F1, TB), TB, 0, stream>>>(ACC, 0.f, (size_t)N * F1);
  k_edge_max<<<cdiv((size_t)E2, TB), TB, 0, stream>>>(ei, E, N, ES, ED, MX, HEADS);
  k_edge_exp<<<cdiv((size_t)E2, TB), TB, 0, stream>>>(ei, E, N, ES, ED, MX, EX1, DN, HEADS);
  k_edge_scatter<<<cdiv((size_t)E2 * HEADS, TB), TB, 0, stream>>>(ei, E, N, Z1, EX1, DN, ACC, HEADS, HIDD);
  k_elu_bias<<<cdiv((size_t)N * F1, TB), TB, 0, stream>>>(ACC, b1, N, F1);  // ACC = h1

  // ================= Layer 2 (1 head, dch=64) =================
  // Z2 = h1 @ W2   [N,256] x [256,64]   (writes into retired Z1 region)
  k_gemm_bf16<HEADS * HIDD><<<dim3(N / 16, 1), 128, 0, stream>>>(ACC, W2, Z2, N, HIDD);
  k_logits<<<cdiv((size_t)N, TB), TB, 0, stream>>>(Z2, as2, ad2, ES, ED, N, 1, HIDD);
  k_fill<<<cdiv((size_t)N, TB), TB, 0, stream>>>(MX, NEG_INF, (size_t)N);
  k_fill<<<cdiv((size_t)N, TB), TB, 0, stream>>>(DN, 0.f, (size_t)N);
  k_fill<<<cdiv((size_t)N * HIDD, TB), TB, 0, stream>>>(ACC2, 0.f, (size_t)N * HIDD);
  k_edge_max<<<cdiv((size_t)E2, TB), TB, 0, stream>>>(ei, E, N, ES, ED, MX, 1);
  k_edge_exp<<<cdiv((size_t)E2, TB), TB, 0, stream>>>(ei, E, N, ES, ED, MX, EX2, DN, 1);
  k_edge_scatter<<<cdiv((size_t)E2, TB), TB, 0, stream>>>(ei, E, N, Z2, EX2, DN, ACC2, 1, HIDD);
  k_elu_bias<<<cdiv((size_t)N * HIDD, TB), TB, 0, stream>>>(ACC2, b2, N, HIDD); // ACC2 = h2

  // ================= Attentional pooling =================
  // G1 = h2 @ Wg1  [N,64] x [64,64]
  k_gemm_bf16<HIDD><<<dim3(N / 16, 1), 128, 0, stream>>>(ACC2, Wg1, G1, N, HIDD);
  k_gate<<<cdiv((size_t)N, TB), TB, 0, stream>>>(G1, bg1, Wg2, bg2, GATE, N, HIDD);
  k_fill<<<1, NGRAPH, 0, stream>>>(GM, NEG_INF, (size_t)NGRAPH);
  k_fill<<<1, NGRAPH, 0, stream>>>(GDEN, 0.f, (size_t)NGRAPH);
  k_fill<<<cdiv((size_t)NGRAPH * HIDD, TB), TB, 0, stream>>>(out, 0.f, (size_t)NGRAPH * HIDD);
  k_bmax<<<cdiv((size_t)N, TB), TB, 0, stream>>>(GATE, bm, GM, N);
  k_bexp<<<cdiv((size_t)N, TB), TB, 0, stream>>>(GATE, bm, GM, EXG, GDEN, N);
  k_bout<<<cdiv((size_t)N * HIDD, TB), TB, 0, stream>>>(ACC2, EXG, GDEN, bm, out, N, HIDD);
}
